// Transformer_50723563766266
// MI455X (gfx1250) — matrix operand order repaired and checked by reading, not compiled, not run on hardware
//
#include <hip/hip_runtime.h>
#include <math.h>

typedef __bf16 bf16;
typedef bf16  v8bf  __attribute__((ext_vector_type(8)));
typedef bf16  v16bf __attribute__((ext_vector_type(16)));
typedef float v8f   __attribute__((ext_vector_type(8)));

#define B_   2
#define S_   2048
#define D_   1024
#define H_   16
#define HD_  64
#define MLPD 4096
#define R_   (B_ * S_)

static __device__ __forceinline__ v8f wmma_bf16(v16bf a, v16bf b, v8f c) {
  // D(f32,16x16) = A(bf16,16x32) * B(bf16,32x16) + C
  return __builtin_amdgcn_wmma_f32_16x16x32_bf16(false, a, false, b, (short)0, c,
                                                 false, false);
}

// ---------------------------------------------------------------------------
// LayerNorm over D_=1024, one row per block, writes bf16 for WMMA consumption.
// ---------------------------------------------------------------------------
__global__ __launch_bounds__(256) void ln_bf16_kernel(const float* __restrict__ x,
                                                      const float* __restrict__ w,
                                                      const float* __restrict__ b,
                                                      bf16* __restrict__ out) {
  __shared__ float red[256];
  const int row = blockIdx.x;
  const int tid = threadIdx.x;
  const size_t base = (size_t)row * D_;
  float v[4];
  float s = 0.f;
#pragma unroll
  for (int j = 0; j < 4; ++j) { v[j] = x[base + tid + j * 256]; s += v[j]; }
  red[tid] = s; __syncthreads();
  for (int st = 128; st > 0; st >>= 1) { if (tid < st) red[tid] += red[tid + st]; __syncthreads(); }
  const float mu = red[0] * (1.0f / D_);
  __syncthreads();
  float s2 = 0.f;
#pragma unroll
  for (int j = 0; j < 4; ++j) { float d = v[j] - mu; s2 += d * d; }
  red[tid] = s2; __syncthreads();
  for (int st = 128; st > 0; st >>= 1) { if (tid < st) red[tid] += red[tid + st]; __syncthreads(); }
  const float rstd = rsqrtf(red[0] * (1.0f / D_) + 1e-5f);
#pragma unroll
  for (int j = 0; j < 4; ++j) {
    int c = tid + j * 256;
    out[base + c] = (bf16)((v[j] - mu) * rstd * w[c] + b[c]);
  }
}

// ---------------------------------------------------------------------------
// Tiled GEMM: C[M,N] = act(A_bf16[M,K] @ B_f32[K,N] + bias) (+ residual).
// BM=128 BN=128 BK=32, 256 threads = 8 waves in a 4(M) x 2(N) grid.
// Each wave computes a 32x64 C tile = 8 v_wmma_f32_16x16x32_bf16 per k-step,
// fed by 2 A-frags + 4 B-frags, all as contiguous ds_load_b128:
//   As row-major, stride 40 (frag: 16 contiguous at +16*half)
//   Bs TRANSPOSED (n-major), stride 48 (frag: 16 contiguous at +16*half)
// Ping-pong double-buffered LDS: iteration i fills buf^1 while WMMAs consume
// buf; one barrier per k-step. Unconditional clamped global_prefetch of the
// k+2 weight tile (weights stay resident in MI455X's 192MB L2).
// ---------------------------------------------------------------------------
#define LDA_S 40
#define LDB_S 48
__global__ __launch_bounds__(256) void gemm_bf16_kernel(
    const bf16* __restrict__ A, int lda,
    const float* __restrict__ Bm, int ldb,
    const float* __restrict__ bias,
    const float* __restrict__ resid,
    float* __restrict__ C32, bf16* __restrict__ C16, int ldc,
    int K, int apply_gelu) {
  __shared__ bf16 As[2][128 * LDA_S];
  __shared__ bf16 Bs[2][128 * LDB_S];
  const int tid = threadIdx.x;
  const int lane = tid & 31;
  const int wid = tid >> 5;
  const int wm = wid >> 1, wn = wid & 1;
  const int half = lane >> 4, lm = lane & 15;
  const int bm = blockIdx.y * 128, bn = blockIdx.x * 128;

  auto fill_tiles = [&](int buf, int k0) {
#pragma unroll
    for (int j = 0; j < 2; ++j) {          // A tile 128x32 bf16, 8 elems/load
      int idx = (tid + j * 256) * 8;
      int r = idx >> 5, c = idx & 31;
      *(v8bf*)&As[buf][r * LDA_S + c] =
          *(const v8bf*)(A + (size_t)(bm + r) * lda + k0 + c);
    }
    int kpre = k0 + 32;                    // clamped (branch-free) prefetch idx
    if (kpre > K - 32) kpre = K - 32;
#pragma unroll
    for (int j = 0; j < 4; ++j) {          // B tile 32x128 fp32 -> bf16, transposed
      int idx = (tid + j * 256) * 4;
      int r = idx >> 7, c = idx & 127;     // r = k, c = n
      float4 f = *(const float4*)(Bm + (size_t)(k0 + r) * ldb + bn + c);
      Bs[buf][(c + 0) * LDB_S + r] = (bf16)f.x;
      Bs[buf][(c + 1) * LDB_S + r] = (bf16)f.y;
      Bs[buf][(c + 2) * LDB_S + r] = (bf16)f.z;
      Bs[buf][(c + 3) * LDB_S + r] = (bf16)f.w;
      __builtin_prefetch(Bm + (size_t)(kpre + r) * ldb + bn + c, 0, 3);
    }
  };

  v8f acc[2][4];
  v8f zero = {};
#pragma unroll
  for (int mi = 0; mi < 2; ++mi)
#pragma unroll
    for (int ni = 0; ni < 4; ++ni) acc[mi][ni] = zero;

  fill_tiles(0, 0);
  __syncthreads();

  for (int k0 = 0; k0 < K; k0 += 32) {
    const int cur = (k0 >> 5) & 1;
    if (k0 + 32 < K) fill_tiles(cur ^ 1, k0 + 32);  // overlap with WMMAs below

    v16bf a[2], b[4];
#pragma unroll
    for (int mi = 0; mi < 2; ++mi) {
      int row = wm * 32 + mi * 16 + lm;
#pragma unroll
      for (int e = 0; e < 16; ++e) {
        // A 16x32 frag: k = 16*half + e   (matches B-frag order)
        int ka = (half << 4) + e;
        a[mi][e] = As[cur][row * LDA_S + ka];
      }
    }
#pragma unroll
    for (int ni = 0; ni < 4; ++ni) {
      int n = wn * 64 + ni * 16 + lm;
#pragma unroll
      for (int e = 0; e < 16; ++e) {
        // B 32x16 frag: k = 16*half + e -> contiguous in transposed Bs
        b[ni][e] = Bs[cur][n * LDB_S + (half << 4) + e];
      }
    }
#pragma unroll
    for (int mi = 0; mi < 2; ++mi)
#pragma unroll
      for (int ni = 0; ni < 4; ++ni)
        acc[mi][ni] = wmma_bf16(a[mi], b[ni], acc[mi][ni]);

    __syncthreads();  // publish next tile / protect buffer reuse
  }

  // C/D frag: row = vgpr + 8*(lane>=16), col = lane&15
#pragma unroll
  for (int mi = 0; mi < 2; ++mi) {
    int row0 = bm + wm * 32 + mi * 16 + half * 8;
#pragma unroll
    for (int r = 0; r < 8; ++r) {
      int row = row0 + r;
#pragma unroll
      for (int ni = 0; ni < 4; ++ni) {
        int col = bn + wn * 64 + ni * 16 + lm;
        float vv = acc[mi][ni][r] + bias[col];
        if (apply_gelu) vv = 0.5f * vv * (1.0f + erff(vv * 0.70710678118f));
        if (resid) vv += resid[(size_t)row * ldc + col];
        if (C32) C32[(size_t)row * ldc + col] = vv;
        if (C16) C16[(size_t)row * ldc + col] = (bf16)vv;
      }
    }
  }
}

// ---------------------------------------------------------------------------
// Flash attention: head_dim=64, S=2048, scale=1/8. Q tile 64 rows, K/V tiles
// of 64 keys streamed through double-buffered LDS (fill kt+1 while computing
// kt). QK^T and PV are bf16 WMMAs; online softmax uses shfl_xor reductions
// over the 16-lane halves matching the C-frag layout. V is stored TRANSPOSED
// in LDS (d-major, stride 72) so PV B-frags are contiguous ds_load_b128;
// K/Q row-major stride 72. Block = 128 threads (4 waves), wave w owns q-rows
// 16w..16w+15.
// ---------------------------------------------------------------------------
#define LDK_S 72
#define NKT   (S_ / 64)
__global__ __launch_bounds__(128) void flash_attn_kernel(
    const bf16* __restrict__ Qp, const bf16* __restrict__ Kp,
    const bf16* __restrict__ Vp, bf16* __restrict__ O) {
  __shared__ bf16 Qs[64 * LDK_S];
  __shared__ bf16 Ks[2][64 * LDK_S];
  __shared__ bf16 Vt[2][64 * LDK_S];  // Vt[d * 72 + k]
  __shared__ bf16 Ps[4 * 16 * 64];    // per-wave P slab for C->A transpose
  const int tid = threadIdx.x;
  const int lane = tid & 31, wid = tid >> 5;
  const int half = lane >> 4, lm = lane & 15;
  const int q0 = blockIdx.x * 64;
  const int h = blockIdx.y;
  const int b = blockIdx.z;
  const size_t base = (size_t)b * S_ * D_ + (size_t)h * HD_;

  auto fill_kv = [&](int buf, int kt) {
#pragma unroll
    for (int j = 0; j < 4; ++j) {
      int idx = (tid + j * 128) * 8;
      int r = idx >> 6, c = idx & 63;         // r = key, c = d
      *(v8bf*)&Ks[buf][r * LDK_S + c] =
          *(const v8bf*)(Kp + base + (size_t)(kt * 64 + r) * D_ + c);
      v8bf vv = *(const v8bf*)(Vp + base + (size_t)(kt * 64 + r) * D_ + c);
#pragma unroll
      for (int t = 0; t < 8; ++t) Vt[buf][(c + t) * LDK_S + r] = vv[t];
    }
  };

#pragma unroll
  for (int j = 0; j < 4; ++j) {
    int idx = (tid + j * 128) * 8;
    int r = idx >> 6, c = idx & 63;
    *(v8bf*)&Qs[r * LDK_S + c] = *(const v8bf*)(Qp + base + (size_t)(q0 + r) * D_ + c);
  }
  fill_kv(0, 0);
  __syncthreads();

  v16bf aq0, aq1;  // Q 16x64 as two 16x32 A-frags (contraction over d)
#pragma unroll
  for (int e = 0; e < 16; ++e) {
    int kl = (half << 4) + e;
    aq0[e] = Qs[(wid * 16 + lm) * LDK_S + kl];
    aq1[e] = Qs[(wid * 16 + lm) * LDK_S + 32 + kl];
  }

  v8f zero = {};
  v8f oacc[4];
  oacc[0] = zero; oacc[1] = zero; oacc[2] = zero; oacc[3] = zero;
  float mrow[8], lrow[8];
#pragma unroll
  for (int r = 0; r < 8; ++r) { mrow[r] = -1e30f; lrow[r] = 0.f; }

  for (int kt = 0; kt < NKT; ++kt) {
    const int cur = kt & 1;
    if (kt + 1 < NKT) fill_kv(cur ^ 1, kt + 1);  // overlap with compute below

    // S = Q @ K^T   (B-frag: n = key index, contraction k = d; contiguous in Ks)
    v8f sacc[4];
    sacc[0] = zero; sacc[1] = zero; sacc[2] = zero; sacc[3] = zero;
#pragma unroll
    for (int nc = 0; nc < 4; ++nc) {
      v16bf bk0, bk1;
#pragma unroll
      for (int e = 0; e < 16; ++e) {
        int d0 = (half << 4) + e;
        bk0[e] = Ks[cur][(nc * 16 + lm) * LDK_S + d0];
        bk1[e] = Ks[cur][(nc * 16 + lm) * LDK_S + 32 + d0];
      }
      sacc[nc] = wmma_bf16(aq0, bk0, sacc[nc]);
      sacc[nc] = wmma_bf16(aq1, bk1, sacc[nc]);
    }
#pragma unroll
    for (int nc = 0; nc < 4; ++nc)
#pragma unroll
      for (int r = 0; r < 8; ++r) sacc[nc][r] *= 0.125f;  // 1/sqrt(64)

    // online softmax per row (row = r + 8*half in this wave's 16-row tile)
#pragma unroll
    for (int r = 0; r < 8; ++r) {
      float mx = fmaxf(fmaxf(sacc[0][r], sacc[1][r]), fmaxf(sacc[2][r], sacc[3][r]));
#pragma unroll
      for (int off = 8; off >= 1; off >>= 1) mx = fmaxf(mx, __shfl_xor(mx, off, 16));
      float mnew = fmaxf(mrow[r], mx);
      float alpha = __expf(mrow[r] - mnew);
      mrow[r] = mnew;
      float ps = 0.f;
#pragma unroll
      for (int nc = 0; nc < 4; ++nc) {
        float p = __expf(sacc[nc][r] - mnew);
        sacc[nc][r] = p;
        ps += p;
      }
#pragma unroll
      for (int off = 8; off >= 1; off >>= 1) ps += __shfl_xor(ps, off, 16);
      lrow[r] = lrow[r] * alpha + ps;
#pragma unroll
      for (int dc = 0; dc < 4; ++dc) oacc[dc][r] *= alpha;
    }

    // transpose P through this wave's LDS slab (C layout -> A layout)
    bf16* Pw = &Ps[wid * 16 * 64];
#pragma unroll
    for (int r = 0; r < 8; ++r) {
      int prow = r + half * 8;
#pragma unroll
      for (int nc = 0; nc < 4; ++nc)
        Pw[prow * 64 + nc * 16 + lm] = (bf16)sacc[nc][r];
    }

    v16bf ap0, ap1;
#pragma unroll
    for (int e = 0; e < 16; ++e) {
      int kl = (half << 4) + e;
      ap0[e] = Pw[lm * 64 + kl];
      ap1[e] = Pw[lm * 64 + 32 + kl];
    }
    // O += P @ V   (B-frag: n = d, contraction k = key; contiguous in Vt)
#pragma unroll
    for (int dc = 0; dc < 4; ++dc) {
      v16bf bv0, bv1;
#pragma unroll
      for (int e = 0; e < 16; ++e) {
        int k0 = (half << 4) + e;
        bv0[e] = Vt[cur][(dc * 16 + lm) * LDK_S + k0];
        bv1[e] = Vt[cur][(dc * 16 + lm) * LDK_S + 32 + k0];
      }
      oacc[dc] = wmma_bf16(ap0, bv0, oacc[dc]);
      oacc[dc] = wmma_bf16(ap1, bv1, oacc[dc]);
    }

    __syncthreads();  // publish next K/V tile / protect buffer reuse
  }

#pragma unroll
  for (int r = 0; r < 8; ++r) {
    float inv = 1.0f / lrow[r];
    int row = q0 + wid * 16 + r + half * 8;
#pragma unroll
    for (int dc = 0; dc < 4; ++dc)
      O[base + (size_t)row * D_ + dc * 16 + lm] = (bf16)(oacc[dc][r] * inv);
  }
}

// ---------------------------------------------------------------------------
// Host orchestration.
// ---------------------------------------------------------------------------
extern "C" void kernel_launch(void* const* d_in, const int* in_sizes, int n_in,
                              void* d_out, int out_size, void* d_ws, size_t ws_size,
                              hipStream_t stream) {
  (void)in_sizes; (void)n_in; (void)out_size; (void)ws_size;
  const float* x     = (const float*)d_in[0];
  const float* ln1_w = (const float*)d_in[1];
  const float* ln1_b = (const float*)d_in[2];
  const float* qkv_w = (const float*)d_in[3];
  const float* qkv_b = (const float*)d_in[4];
  const float* in_w  = (const float*)d_in[5];
  const float* in_b  = (const float*)d_in[6];
  const float* out_w = (const float*)d_in[7];
  const float* out_b = (const float*)d_in[8];
  const float* ln2_w = (const float*)d_in[9];
  const float* ln2_b = (const float*)d_in[10];
  const float* w1    = (const float*)d_in[11];
  const float* b1    = (const float*)d_in[12];
  const float* w2    = (const float*)d_in[13];
  const float* b2    = (const float*)d_in[14];

  char* wsp = (char*)d_ws;
  size_t off = 0;
  auto alloc = [&](size_t bytes) -> void* {
    void* p = wsp + off;
    off += (bytes + 255) & ~(size_t)255;
    return p;
  };
  float* xa = (float*)alloc((size_t)R_ * D_ * 4);        // residual stream A
  float* xb = (float*)alloc((size_t)R_ * D_ * 4);        // residual stream B
  bf16* hb  = (bf16*)alloc((size_t)R_ * D_ * 2);         // LN output
  bf16* big = (bf16*)alloc((size_t)R_ * MLPD * 2);       // qkv (3D cols) / MLP hidden
  bf16* qp  = (bf16*)alloc((size_t)R_ * D_ * 2);
  bf16* kp  = (bf16*)alloc((size_t)R_ * D_ * 2);
  bf16* vp  = (bf16*)alloc((size_t)R_ * D_ * 2);
  bf16* ob  = (bf16*)alloc((size_t)R_ * D_ * 2);

  const float* xin = x;
  for (int i = 0; i < 2; ++i) {
    const float* l_ln1w = ln1_w + (size_t)i * D_;
    const float* l_ln1b = ln1_b + (size_t)i * D_;
    const float* l_qkvw = qkv_w + (size_t)i * D_ * 3 * D_;
    const float* l_qkvb = qkv_b + (size_t)i * 3 * D_;
    const float* l_inw  = in_w  + (size_t)i * D_ * 3 * D_;
    const float* l_inb  = in_b  + (size_t)i * 3 * D_;
    const float* l_outw = out_w + (size_t)i * D_ * D_;
    const float* l_outb = out_b + (size_t)i * D_;
    const float* l_ln2w = ln2_w + (size_t)i * D_;
    const float* l_ln2b = ln2_b + (size_t)i * D_;
    const float* l_w1   = w1 + (size_t)i * D_ * MLPD;
    const float* l_b1   = b1 + (size_t)i * MLPD;
    const float* l_w2   = w2 + (size_t)i * MLPD * D_;
    const float* l_b2   = b2 + (size_t)i * D_;

    // h = LN1(x)
    ln_bf16_kernel<<<R_, 256, 0, stream>>>(xin, l_ln1w, l_ln1b, hb);
    // qkv = h @ qkv_w + qkv_b
    gemm_bf16_kernel<<<dim3(3 * D_ / 128, R_ / 128), 256, 0, stream>>>(
        hb, D_, l_qkvw, 3 * D_, l_qkvb, nullptr, nullptr, big, 3 * D_, D_, 0);
    // qp/kp/vp = {q,k,v} @ in_w[:, cD:(c+1)D] + in_b[cD:(c+1)D]
    for (int c = 0; c < 3; ++c) {
      bf16* dst = (c == 0) ? qp : (c == 1) ? kp : vp;
      gemm_bf16_kernel<<<dim3(D_ / 128, R_ / 128), 256, 0, stream>>>(
          big + c * D_, 3 * D_, l_inw + c * D_, 3 * D_, l_inb + c * D_,
          nullptr, nullptr, dst, D_, D_, 0);
    }
    // o = softmax(qp kp^T / 8) vp
    flash_attn_kernel<<<dim3(S_ / 64, H_, B_), 128, 0, stream>>>(qp, kp, vp, ob);
    // x = o @ out_w + out_b + x
    gemm_bf16_kernel<<<dim3(D_ / 128, R_ / 128), 256, 0, stream>>>(
        ob, D_, l_outw, D_, l_outb, xin, xa, nullptr, D_, D_, 0);
    // h = LN2(x)
    ln_bf16_kernel<<<R_, 256, 0, stream>>>(xa, l_ln2w, l_ln2b, hb);
    // h1 = gelu(h @ w1 + b1)
    gemm_bf16_kernel<<<dim3(MLPD / 128, R_ / 128), 256, 0, stream>>>(
        hb, D_, l_w1, MLPD, l_b1, nullptr, nullptr, big, MLPD, D_, 1);
    // x = h1 @ w2 + b2 + x   (final layer writes d_out directly)
    float* xout = (i == 1) ? (float*)d_out : xb;
    gemm_bf16_kernel<<<dim3(D_ / 128, R_ / 128), 256, 0, stream>>>(
        big, MLPD, l_w2, D_, l_b2, xa, xout, nullptr, D_, MLPD, 0);
    xin = xb;
  }
}
